// FisrtFrame_Sparse4DHead_83305185673271
// MI455X (gfx1250) — matrix operand order repaired and checked by reading, not compiled, not run on hardware
//
#include <hip/hip_runtime.h>
#include <hip/hip_bf16.h>

typedef __attribute__((ext_vector_type(16))) _Float16 v16h;
typedef __attribute__((ext_vector_type(8)))  _Float16 v8h;
typedef __attribute__((ext_vector_type(8)))  float    v8f;

#define C_DIM      256
#define NA         900          // anchors
#define M_PAD      912          // 57 * 16
#define NCAMS      6
#define NLVL       4
#define NPTS       13
#define NGRP       8
#define NW         2496         // 6*4*13*8
#define NI         312          // 6*4*13
#define SUM_HW     14960

// ---- workspace layout (bytes) ----
constexpr size_t WS_Q      = 0;                                          // 912*256 f16
constexpr size_t WS_WFC16  = WS_Q      + (size_t)M_PAD * C_DIM * 2;      // 256*2496 f16
constexpr size_t WS_OUTW16 = WS_WFC16  + (size_t)C_DIM * NW * 2;         // 256*256 f16
constexpr size_t WS_W      = WS_OUTW16 + (size_t)C_DIM * C_DIM * 2;      // 900*2496 f32
constexpr size_t WS_PTS    = WS_W      + (size_t)NA * NW * 4;            // 900*78 float4
constexpr size_t WS_AGG16  = WS_PTS    + (size_t)NA * NCAMS * NPTS * 16; // 912*256 f16

// =====================================================================
// Kernel 1: anchor embed + Q(f16), learned keypoints, projection
// grid = M_PAD blocks x 256 threads (one block per anchor row incl. pad)
// =====================================================================
__global__ void prep_kernel(const float* __restrict__ inst,
                            const float* __restrict__ anchor,
                            const float* __restrict__ l2i,
                            const float* __restrict__ img_wh,
                            const float* __restrict__ enc_w,
                            const float* __restrict__ enc_b,
                            const float* __restrict__ kp_fixed,
                            const float* __restrict__ kp_w,
                            const float* __restrict__ kp_b,
                            _Float16* __restrict__ q16,
                            float4*  __restrict__ pts)
{
    const int n   = blockIdx.x;
    const int tid = threadIdx.x;
    if (n >= NA) {                       // zero the M-padding rows of Q
        q16[(size_t)n * C_DIM + tid] = (_Float16)0.0f;
        return;
    }

    __shared__ float sA[11];
    __shared__ float sInst[C_DIM];
    __shared__ float sLearn[18];
    __shared__ float sKP[NPTS][3];

    if (tid < 11) sA[tid] = anchor[n * 11 + tid];
    const float iv = inst[(size_t)n * C_DIM + tid];
    sInst[tid] = iv;
    __syncthreads();

    // anchor embedding (11-term GEMV) and Q = inst + embed  (f16)
    float emb = enc_b[tid];
#pragma unroll
    for (int j = 0; j < 11; ++j) emb += sA[j] * enc_w[j * C_DIM + tid];
    q16[(size_t)n * C_DIM + tid] = (_Float16)(iv + emb);

    // learned keypoint offsets: (1x256) @ (256x18)
    if (tid < 18) {
        float acc = kp_b[tid];
        for (int c = 0; c < C_DIM; ++c) acc += sInst[c] * kp_w[c * 18 + tid];
        sLearn[tid] = acc;
    }
    __syncthreads();

    // keypoints: scale by exp(size), rotate by yaw, translate by center
    if (tid < NPTS) {
        const float sx = expf(sA[3]), sy = expf(sA[4]), sz = expf(sA[5]);
        float ox, oy, oz;
        if (tid < 7) { ox = kp_fixed[tid*3+0]; oy = kp_fixed[tid*3+1]; oz = kp_fixed[tid*3+2]; }
        else         { ox = sLearn[(tid-7)*3+0]; oy = sLearn[(tid-7)*3+1]; oz = sLearn[(tid-7)*3+2]; }
        ox *= sx; oy *= sy; oz *= sz;
        const float si = sA[6], co = sA[7];
        sKP[tid][0] = co*ox - si*oy + sA[0];
        sKP[tid][1] = si*ox + co*oy + sA[1];
        sKP[tid][2] = oz           + sA[2];
    }
    __syncthreads();

    // project to all cameras: 6 cams x 13 pts = 78 threads
    if (tid < NCAMS * NPTS) {
        const int cam = tid / NPTS, pt = tid % NPTS;
        const float* M = l2i + cam * 16;
        const float x = sKP[pt][0], y = sKP[pt][1], z = sKP[pt][2];
        const float p0 = M[0]*x + M[1]*y + M[2] *z + M[3];
        const float p1 = M[4]*x + M[5]*y + M[6] *z + M[7];
        const float p2 = M[8]*x + M[9]*y + M[10]*z + M[11];
        const float d  = fmaxf(p2, 1e-5f);
        const float px = (p0 / d) / img_wh[cam*2 + 0];
        const float py = (p1 / d) / img_wh[cam*2 + 1];
        pts[((size_t)n * NCAMS + cam) * NPTS + pt] =
            make_float4(px, py, (p2 > 1e-5f) ? 1.0f : 0.0f, 0.0f);
    }
}

// =====================================================================
// Kernel 2: weight matrices f32 -> f16, zero agg pad rows
// =====================================================================
__global__ void cvt_kernel(const float* __restrict__ wfc_w,
                           const float* __restrict__ out_w,
                           _Float16* __restrict__ wfc16,
                           _Float16* __restrict__ outw16,
                           _Float16* __restrict__ agg16)
{
    const int idx = blockIdx.x * blockDim.x + threadIdx.x;
    if (idx < C_DIM * NW)    wfc16[idx]  = (_Float16)wfc_w[idx];
    if (idx < C_DIM * C_DIM) outw16[idx] = (_Float16)out_w[idx];
    if (idx < (M_PAD - NA) * C_DIM)
        agg16[(size_t)NA * C_DIM + idx] = (_Float16)0.0f;
}

// =====================================================================
// WMMA 16x64 wave tile: D[j](16x16 f32) = A(16xK f16) x B[j](Kx16 f16),
// j = 0..3 over four adjacent N tiles. A fragment is loaded once per
// K-step and reused by 4 WMMAs (A: 0.5 b128 loads per WMMA vs 2 unblocked).
// CDNA5 16-bit layouts: A lane=M with split-K halves; B lane=N column, 16
// halves = K rows -> each B fetch is 16 halves strided by ldb, and the
// 4 N-tiles span one contiguous 128B line per half-wave per K row.
// =====================================================================
__device__ __forceinline__ void wmma_tile_f16_n4(const _Float16* __restrict__ A, int lda,
                                                 const _Float16* __restrict__ Bm, int ldb,
                                                 int K, v8f acc[4])
{
    const int  lane = threadIdx.x & 31;
    const bool hiHalf = lane >= 16;
    const _Float16* arow = A + (size_t)(lane & 15) * lda;
    for (int k0 = 0; k0 < K; k0 += 32) {
        if (k0 + 32 < K)
            __builtin_prefetch(Bm + (size_t)(k0 + 32 + lane) * ldb, 0, 1);
        // A: lanes 0-15 hold K = k0+[0..7] & k0+[16..23]; lanes 16-31 hold +8 / +24
        const v8h lo = *(const v8h*)(arow + k0 + (hiHalf ? 8  : 0));
        const v8h hh = *(const v8h*)(arow + k0 + (hiHalf ? 24 : 16));
        v16h a;
#pragma unroll
        for (int i = 0; i < 8; ++i) { a[i] = lo[i]; a[i + 8] = hh[i]; }
        const _Float16* bcol = Bm + (size_t)k0 * ldb + (lane & 15);
#pragma unroll
        for (int j = 0; j < 4; ++j) {
            // B: lane holds column N = lane&15, 16 halves are K in the same order as A
            v16h b;
#pragma unroll
            for (int i = 0; i < 8; ++i) {
                b[i]     = bcol[(size_t)((hiHalf ? 8  : 0)  + i) * ldb + j * 16];
                b[i + 8] = bcol[(size_t)((hiHalf ? 24 : 16) + i) * ldb + j * 16];
            }
            acc[j] = __builtin_amdgcn_wmma_f32_16x16x32_f16(
                         false, a, false, b, (short)0, acc[j], false, false);
        }
    }
}

// =====================================================================
// Kernel 3: W = Q(912x256) @ wfc16(256x2496) + wfc_b   -> f32
// grid = (39 nblock(x4 tiles), 57 mtile), block = 32 (one wave)
// =====================================================================
__global__ void gemm_w_kernel(const _Float16* __restrict__ q16,
                              const _Float16* __restrict__ wfc16,
                              const float* __restrict__ wfc_b,
                              float* __restrict__ W)
{
    const int nb = blockIdx.x, mt = blockIdx.y;
    const int lane = threadIdx.x & 31;
    v8f acc[4] = {};
    wmma_tile_f16_n4(q16 + (size_t)mt * 16 * C_DIM, C_DIM,
                     wfc16 + nb * 64, NW, C_DIM, acc);
    const int mbase = mt * 16 + (lane >= 16 ? 8 : 0);
#pragma unroll
    for (int j = 0; j < 4; ++j) {
        const int col  = nb * 64 + j * 16 + (lane & 15);
        const float bias = wfc_b[col];
#pragma unroll
        for (int r = 0; r < 8; ++r) {
            const int m = mbase + r;
            if (m < NA) W[(size_t)m * NW + col] = acc[j][r] + bias;
        }
    }
}

// =====================================================================
// Kernel 4: softmax over the 312 (cam,level,pt) entries per (anchor,group)
// block = 256 = 8 waves; wave g owns group g (wave32 shuffle reductions)
// =====================================================================
__global__ void softmax_kernel(float* __restrict__ W)
{
    const int n   = blockIdx.x;
    const int tid = threadIdx.x;
    __shared__ float sv[NW];
    float* row = W + (size_t)n * NW;
    for (int i = tid; i < NW; i += 256) sv[i] = row[i];
    __syncthreads();

    const int g = tid >> 5;      // wave id == group id
    const int l = tid & 31;
    float mx = -3.4e38f;
    for (int i = l; i < NI; i += 32) mx = fmaxf(mx, sv[i * NGRP + g]);
#pragma unroll
    for (int o = 16; o > 0; o >>= 1) mx = fmaxf(mx, __shfl_xor(mx, o, 32));
    float sum = 0.0f;
    for (int i = l; i < NI; i += 32) {
        const float e = expf(sv[i * NGRP + g] - mx);
        sv[i * NGRP + g] = e;
        sum += e;
    }
#pragma unroll
    for (int o = 16; o > 0; o >>= 1) sum += __shfl_xor(sum, o, 32);
    const float inv = 1.0f / sum;
    for (int i = l; i < NI; i += 32) row[i * NGRP + g] = sv[i * NGRP + g] * inv;
}

// =====================================================================
// Kernel 5: bilinear sample + weighted aggregation
// block per anchor, thread per channel (coalesced 1KB tap rows, L2-resident)
// =====================================================================
__global__ void sample_kernel(const float* __restrict__ feat,
                              const float* __restrict__ W,
                              const float4* __restrict__ pts,
                              _Float16* __restrict__ agg16)
{
    const int n = blockIdx.x;
    const int c = threadIdx.x;
    const int g = c >> 5;                     // channel group (32 ch each)
    __shared__ float  sW[NW];
    __shared__ float4 sP[NCAMS * NPTS];
    for (int i = c; i < NW; i += 256) sW[i] = W[(size_t)n * NW + i];
    if (c < NCAMS * NPTS) sP[c] = pts[(size_t)n * NCAMS * NPTS + c];
    __syncthreads();

    constexpr int Hs[4] = {64, 32, 16, 8};
    constexpr int Ws_[4] = {176, 88, 44, 22};
    constexpr int St[4] = {0, 11264, 14080, 14784};

    float acc = 0.0f;
    for (int cam = 0; cam < NCAMS; ++cam) {
        const float* fcam = feat + (size_t)cam * SUM_HW * C_DIM + c;
        for (int pt = 0; pt < NPTS; ++pt) {
            const float4 P = sP[cam * NPTS + pt];
            if (P.z == 0.0f) continue;        // behind camera -> all taps zero
#pragma unroll
            for (int l = 0; l < 4; ++l) {
                const int Hh = Hs[l], Ww = Ws_[l], st = St[l];
                const float x = P.x * (float)Ww - 0.5f;
                const float y = P.y * (float)Hh - 0.5f;
                const float xf = floorf(x), yf = floorf(y);
                const int x0 = (int)xf, y0 = (int)yf;
                const float wx = x - xf, wy = y - yf;
                const float wv = sW[((cam * 4 + l) * NPTS + pt) * NGRP + g];
                const bool okx0 = (x0 >= 0) && (x0 < Ww);
                const bool okx1 = (x0 + 1 >= 0) && (x0 + 1 < Ww);
                const bool oky0 = (y0 >= 0) && (y0 < Hh);
                const bool oky1 = (y0 + 1 >= 0) && (y0 + 1 < Hh);
                float v = 0.0f;
                if (okx0 && oky0) v += (1.f-wx)*(1.f-wy) * fcam[(size_t)(st + y0*Ww + x0)     * C_DIM];
                if (okx1 && oky0) v +=      wx *(1.f-wy) * fcam[(size_t)(st + y0*Ww + x0 + 1) * C_DIM];
                if (okx0 && oky1) v += (1.f-wx)*     wy  * fcam[(size_t)(st + (y0+1)*Ww + x0)     * C_DIM];
                if (okx1 && oky1) v +=      wx *     wy  * fcam[(size_t)(st + (y0+1)*Ww + x0 + 1) * C_DIM];
                acc += wv * v;
            }
        }
    }
    agg16[(size_t)n * C_DIM + c] = (_Float16)acc;
}

// =====================================================================
// Kernel 6: out = agg(912x256 f16) @ out_w(256x256 f16) + out_b + inst
// grid = (4 nblock(x4 tiles), 57 mtile), block = 32
// =====================================================================
__global__ void gemm_out_kernel(const _Float16* __restrict__ agg16,
                                const _Float16* __restrict__ outw16,
                                const float* __restrict__ out_b,
                                const float* __restrict__ inst,
                                float* __restrict__ out)
{
    const int nb = blockIdx.x, mt = blockIdx.y;
    const int lane = threadIdx.x & 31;
    v8f acc[4] = {};
    wmma_tile_f16_n4(agg16 + (size_t)mt * 16 * C_DIM, C_DIM,
                     outw16 + nb * 64, C_DIM, C_DIM, acc);
    const int mbase = mt * 16 + (lane >= 16 ? 8 : 0);
#pragma unroll
    for (int j = 0; j < 4; ++j) {
        const int col  = nb * 64 + j * 16 + (lane & 15);
        const float bias = out_b[col];
#pragma unroll
        for (int r = 0; r < 8; ++r) {
            const int m = mbase + r;
            if (m < NA)
                out[(size_t)m * C_DIM + col] =
                    acc[j][r] + bias + inst[(size_t)m * C_DIM + col];
        }
    }
}

// =====================================================================
extern "C" void kernel_launch(void* const* d_in, const int* in_sizes, int n_in,
                              void* d_out, int out_size, void* d_ws, size_t ws_size,
                              hipStream_t stream)
{
    const float* inst     = (const float*)d_in[0];
    const float* anchor   = (const float*)d_in[1];
    // d_in[2] time_interval: unused by reference
    const float* feature  = (const float*)d_in[3];
    // d_in[4] spatial_shapes, d_in[5] level_start_index: compile-time constants
    const float* l2i      = (const float*)d_in[6];
    const float* img_wh   = (const float*)d_in[7];
    const float* enc_w    = (const float*)d_in[8];
    const float* enc_b    = (const float*)d_in[9];
    const float* kp_fixed = (const float*)d_in[10];
    const float* kp_w     = (const float*)d_in[11];
    const float* kp_b     = (const float*)d_in[12];
    const float* wfc_w    = (const float*)d_in[13];
    const float* wfc_b    = (const float*)d_in[14];
    const float* out_w    = (const float*)d_in[15];
    const float* out_b    = (const float*)d_in[16];

    char* ws = (char*)d_ws;
    _Float16* q16    = (_Float16*)(ws + WS_Q);
    _Float16* wfc16  = (_Float16*)(ws + WS_WFC16);
    _Float16* outw16 = (_Float16*)(ws + WS_OUTW16);
    float*    Wbuf   = (float*)   (ws + WS_W);
    float4*   ptsbuf = (float4*)  (ws + WS_PTS);
    _Float16* agg16  = (_Float16*)(ws + WS_AGG16);

    prep_kernel<<<M_PAD, 256, 0, stream>>>(inst, anchor, l2i, img_wh,
                                           enc_w, enc_b, kp_fixed, kp_w, kp_b,
                                           q16, ptsbuf);
    cvt_kernel<<<(C_DIM * NW + 255) / 256, 256, 0, stream>>>(wfc_w, out_w,
                                                             wfc16, outw16, agg16);
    gemm_w_kernel<<<dim3(NW / 64, M_PAD / 16), 32, 0, stream>>>(q16, wfc16,
                                                                wfc_b, Wbuf);
    softmax_kernel<<<NA, 256, 0, stream>>>(Wbuf);
    sample_kernel<<<NA, 256, 0, stream>>>(feature, Wbuf, ptsbuf, agg16);
    gemm_out_kernel<<<dim3(C_DIM / 64, M_PAD / 16), 32, 0, stream>>>(
        agg16, outw16, out_b, inst, (float*)d_out);
}
